// SkillDecoder_61684320305864
// MI455X (gfx1250) — matrix operand order repaired and checked by reading, not compiled, not run on hardware
//
#include <hip/hip_runtime.h>

// ---------------------------------------------------------------------------
// LSTM decoder for MI455X (gfx1250, wave32, WMMA).
// B=2048, Z=64, A=8, H=512, T=64. Dominant cost: 64 serial GEMMs
// h(2048x512) @ W_hh^T(512x2048) -> f16 WMMA, f32 accumulate, fused LSTM cell.
// All working data is L2-resident (<30MB), so design for matrix-op throughput.
// ---------------------------------------------------------------------------

typedef __attribute__((ext_vector_type(16))) _Float16 v16h;
typedef __attribute__((ext_vector_type(8)))  _Float16 v8h;
typedef __attribute__((ext_vector_type(8)))  float    v8f;
typedef int v4i_gl __attribute__((__vector_size__(4 * sizeof(int))));  // builtin's pointee type

#define B_   2048
#define H_   512
#define G4H  2048   // 4*H
#define T_   64
#define A_   8
#define Z_   64
#define AZ_  72     // A+Z

// CDNA5 async global->LDS path (bypasses VGPRs, tracked by ASYNCcnt)
#if defined(__gfx1250__) && __has_builtin(__builtin_amdgcn_global_load_async_to_lds_b128)
#define USE_ASYNC 1
#else
#define USE_ASYNC 0
#endif

__device__ __forceinline__ void wait_async0() {
#if __has_builtin(__builtin_amdgcn_s_wait_asynccnt)
    __builtin_amdgcn_s_wait_asynccnt(0);
#else
    asm volatile("s_wait_asynccnt 0x0" ::: "memory");
#endif
}

__device__ __forceinline__ float fast_sig(float x)  { return 1.f / (1.f + __expf(-x)); }
__device__ __forceinline__ float fast_tanh(float x) { return 2.f / (1.f + __expf(-2.f * x)) - 1.f; }

// ---------------------------------------------------------------------------
// Convert W_hh f32 -> f16 and zero h0 (f16) / c0 (f32).
// Both arrays are exactly 4H*H == B*H == 1,048,576 elements.
// ---------------------------------------------------------------------------
__global__ void k_init(const float* __restrict__ W_hh, _Float16* __restrict__ Whh16,
                       _Float16* __restrict__ h0, float* __restrict__ c0) {
    const int i = blockIdx.x * 256 + threadIdx.x;   // grid sized exactly
    Whh16[i] = (_Float16)W_hh[i];
    h0[i]    = (_Float16)0.f;
    c0[i]    = 0.f;
}

// ---------------------------------------------------------------------------
// x_proj[b][n] = b_ih[n] + b_hh[n] + sum_k z[b][k] * W_ih[n][A_+k]
// (first A_ columns of dec_in are zeros -> skipped). One-time cost.
// ---------------------------------------------------------------------------
__global__ void k_xproj(const float* __restrict__ z, const float* __restrict__ W_ih,
                        const float* __restrict__ b_ih, const float* __restrict__ b_hh,
                        float* __restrict__ xp) {
    __shared__ float zs[8][Z_];
    const int tid = threadIdx.x;
    const int n0  = (blockIdx.x & 7) * 256;
    const int b0  = (blockIdx.x >> 3) * 8;
    for (int i = tid; i < 8 * Z_; i += 256)
        zs[i >> 6][i & 63] = z[(size_t)(b0 + (i >> 6)) * Z_ + (i & 63)];
    __syncthreads();

    const int n = n0 + tid;
    const float* wr = W_ih + (size_t)n * AZ_ + A_;
    const float bias = b_ih[n] + b_hh[n];
    float acc[8];
#pragma unroll
    for (int i = 0; i < 8; ++i) acc[i] = bias;
    for (int k = 0; k < Z_; ++k) {
        const float wv = wr[k];
#pragma unroll
        for (int i = 0; i < 8; ++i) acc[i] += zs[i][k] * wv;
    }
#pragma unroll
    for (int i = 0; i < 8; ++i) xp[(size_t)(b0 + i) * G4H + n] = acc[i];
}

// ---------------------------------------------------------------------------
// One LSTM timestep, fully fused:
//   gates = x_proj + h_in @ W_hh^T   (f16 WMMA, f32 acc)
//   c     = sig(f)*c + sig(i)*tanh(g);  h_out = sig(o)*tanh(c)
//
// Grid (16,16): blockIdx.x -> 32 h-columns, blockIdx.y -> 128 batch rows.
// Each block computes gate columns {j0, j0+H, j0+2H, j0+3H} (128 N cols) so
// the cell update is wave-local. 8 waves; wave w owns batch rows rm..rm+15
// and all 8 N-tiles -> A fragment reused 8x per K-chunk.
//
// B tile staged in LDS double-buffered via async global->LDS (no VGPR
// round-trip); pipeline: wait_asynccnt(0) -> barrier -> issue next chunk ->
// compute. All 8 B fragments are preloaded per chunk so the 8 WMMAs issue
// back-to-back instead of stalling on dscnt==0 per tile. Padded LDS stride
// (40 halves) keeps fragment reads bank-conflict-free per half-wave.
// ---------------------------------------------------------------------------
__global__ __launch_bounds__(256)
void k_step(const _Float16* __restrict__ h_in,
            const _Float16* __restrict__ Whh16,
            const float*    __restrict__ xp,
            float*          __restrict__ c,
            _Float16*       __restrict__ h_out) {
    constexpr int LDSTR = 40;  // halves per LDS B row (32 data + 8 pad)
    __shared__ __align__(16) _Float16 ldsB[2][128 * LDSTR];

    const int tid  = threadIdx.x;
    const int wave = tid >> 5;
    const int lane = tid & 31;
    const int l16  = lane & 15;
    const int hsel = lane >> 4;          // half-wave select

    const int r0 = blockIdx.y * 128;     // batch-row base
    const int j0 = blockIdx.x * 32;      // h-column base
    const int rm = r0 + wave * 16;       // this wave's 16 batch rows

    // --- B-tile fill addressing: two 16B chunks per thread per K-chunk ---
    const int c0i = tid, c1i = tid + 256;
    const int r0i = c0i >> 2, k0i = (c0i & 3) * 8;
    const int r1i = c1i >> 2, k1i = (c1i & 3) * 8;
    // LDS row (0..127) -> W_hh row: gate = row>>5, col-in-gate = row&31
    const size_t g0 = (size_t)(((r0i >> 5) << 9) + j0 + (r0i & 31)) * H_ + k0i;
    const size_t g1 = (size_t)(((r1i >> 5) << 9) + j0 + (r1i & 31)) * H_ + k1i;
    const int l0 = r0i * LDSTR + k0i;
    const int l1 = r1i * LDSTR + k1i;

#if USE_ASYNC
    auto asyncFill = [&](int bsel, int kk) {
        __builtin_amdgcn_global_load_async_to_lds_b128(
            (v4i_gl*)(Whh16 + g0 + kk),
            (__attribute__((address_space(3))) v4i_gl*)(&ldsB[bsel][l0]), 0, 0);
        __builtin_amdgcn_global_load_async_to_lds_b128(
            (v4i_gl*)(Whh16 + g1 + kk),
            (__attribute__((address_space(3))) v4i_gl*)(&ldsB[bsel][l1]), 0, 0);
    };
#endif

    // --- A fragment base: lane row rm+l16; half-wave K offset (ISA layout) ---
    const _Float16* aRow = h_in + (size_t)(rm + l16) * H_ + hsel * 8;

    v8f acc[8] = {};

    // prologue prefetch (K-chunk 0)
#if USE_ASYNC
    asyncFill(0, 0);
#else
    uint4 pb0 = *(const uint4*)(Whh16 + g0);
    uint4 pb1 = *(const uint4*)(Whh16 + g1);
#endif
    v8h pa0 = *(const v8h*)(aRow);
    v8h pa1 = *(const v8h*)(aRow + 16);

    for (int kc = 0; kc < 16; ++kc) {       // K = 512 in chunks of 32
        _Float16* buf = ldsB[kc & 1];
#if USE_ASYNC
        wait_async0();                      // own async fills for chunk kc done
        __syncthreads();                    // all waves' fills visible; prev buf free
        if (kc < 15) asyncFill((kc + 1) & 1, (kc + 1) * 32);  // hide under compute
#else
        *(uint4*)(buf + l0) = pb0;
        *(uint4*)(buf + l1) = pb1;
        if (kc < 15) {
            const int kk = (kc + 1) * 32;
            pb0 = *(const uint4*)(Whh16 + g0 + kk);
            pb1 = *(const uint4*)(Whh16 + g1 + kk);
        }
        __syncthreads();
#endif
        const v16h afrag = __builtin_shufflevector(pa0, pa1,
            0,1,2,3,4,5,6,7,8,9,10,11,12,13,14,15);
        if (kc < 15) {                      // A prefetch under compute
            const int kk = (kc + 1) * 32;
            pa0 = *(const v8h*)(aRow + kk);
            pa1 = *(const v8h*)(aRow + kk + 16);
        }

        // preload all 8 B fragments, then issue 8 WMMAs back-to-back
        v8h bl[8], bh[8];
#pragma unroll
        for (int nt = 0; nt < 8; ++nt) {
            const _Float16* bp = buf + (nt * 16 + l16) * LDSTR + hsel * 8;
            bl[nt] = *(const v8h*)(bp);
            bh[nt] = *(const v8h*)(bp + 16);
        }
#pragma unroll
        for (int nt = 0; nt < 8; ++nt) {
            const v16h bfrag = __builtin_shufflevector(bl[nt], bh[nt],
                0,1,2,3,4,5,6,7,8,9,10,11,12,13,14,15);
            acc[nt] = __builtin_amdgcn_wmma_f32_16x16x32_f16(
                false, afrag, false, bfrag, (short)0, acc[nt], false, false);
        }
    }

    // --- fused LSTM cell update (acc[2g+s][v] = gate g, subtile s, row v) ---
#pragma unroll
    for (int s = 0; s < 2; ++s) {
        const int j = j0 + s * 16 + l16;            // h column
#pragma unroll
        for (int v = 0; v < 8; ++v) {
            const int row = rm + v + hsel * 8;      // batch row (C/D layout)
            const size_t xb = (size_t)row * G4H + j;
            const float gi = acc[0 + s][v] + xp[xb];
            const float gf = acc[2 + s][v] + xp[xb + H_];
            const float gg = acc[4 + s][v] + xp[xb + 2 * H_];
            const float go = acc[6 + s][v] + xp[xb + 3 * H_];
            const size_t ci = (size_t)row * H_ + j;
            const float cn = fast_sig(gf) * c[ci] + fast_sig(gi) * fast_tanh(gg);
            c[ci] = cn;
            h_out[ci] = (_Float16)(fast_sig(go) * fast_tanh(cn));
        }
    }
}

// ---------------------------------------------------------------------------
// out[b][t][:] = h_t[b] @ W_out^T + b_out.  One wave per batch row; lanes
// split K, shuffle-reduce. Tiny (1 GFLOP total over all steps).
// ---------------------------------------------------------------------------
__global__ void k_outproj(const _Float16* __restrict__ h, const float* __restrict__ W_out,
                          const float* __restrict__ b_out, float* __restrict__ out, int t) {
    const int wave = threadIdx.x >> 5, lane = threadIdx.x & 31;
    const int b = blockIdx.x * 8 + wave;
    const _Float16* hp = h + (size_t)b * H_;
    float hv[16];
#pragma unroll
    for (int i = 0; i < 16; ++i) hv[i] = (float)hp[lane + i * 32];
#pragma unroll
    for (int a = 0; a < A_; ++a) {
        const float* wp = W_out + a * H_;
        float s = 0.f;
#pragma unroll
        for (int i = 0; i < 16; ++i) s += hv[i] * wp[lane + i * 32];
        for (int off = 16; off; off >>= 1) s += __shfl_down(s, off, 32);
        if (lane == 0) out[(size_t)b * (T_ * A_) + t * A_ + a] = s + b_out[a];
    }
}

// ---------------------------------------------------------------------------
extern "C" void kernel_launch(void* const* d_in, const int* in_sizes, int n_in,
                              void* d_out, int out_size, void* d_ws, size_t ws_size,
                              hipStream_t stream) {
    const float* z     = (const float*)d_in[0];
    const float* W_ih  = (const float*)d_in[1];
    const float* W_hh  = (const float*)d_in[2];
    const float* b_ih  = (const float*)d_in[3];
    const float* b_hh  = (const float*)d_in[4];
    const float* W_out = (const float*)d_in[5];
    const float* b_out = (const float*)d_in[6];
    float* out = (float*)d_out;

    // workspace carve (~26 MB, all sizes 16B-aligned)
    char* ws = (char*)d_ws;
    _Float16* Whh16 = (_Float16*)ws;  ws += (size_t)G4H * H_ * sizeof(_Float16);  //  2 MB
    float*    xp    = (float*)ws;     ws += (size_t)B_  * G4H * sizeof(float);    // 16 MB
    float*    c     = (float*)ws;     ws += (size_t)B_  * H_  * sizeof(float);    //  4 MB
    _Float16* hbuf0 = (_Float16*)ws;  ws += (size_t)B_  * H_  * sizeof(_Float16); //  2 MB
    _Float16* hbuf1 = (_Float16*)ws;                                              //  2 MB

    k_init<<<(G4H * H_) / 256, 256, 0, stream>>>(W_hh, Whh16, hbuf0, c);
    k_xproj<<<(B_ / 8) * (G4H / 256), 256, 0, stream>>>(z, W_ih, b_ih, b_hh, xp);

    for (int t = 0; t < T_; ++t) {
        _Float16* hin  = (t & 1) ? hbuf1 : hbuf0;
        _Float16* hout = (t & 1) ? hbuf0 : hbuf1;
        k_step<<<dim3(H_ / 32, B_ / 128), 256, 0, stream>>>(hin, Whh16, xp, c, hout);
        k_outproj<<<B_ / 8, 256, 0, stream>>>(hout, W_out, b_out, out, t);
    }
}
